// GPLLA_5729486372996
// MI455X (gfx1250) — matrix operand order repaired and checked by reading, not compiled, not run on hardware
//
#include <hip/hip_runtime.h>
#include <hip/hip_bf16.h>

// ---------------------------------------------------------------------------
// GP last-layer-approx posterior: mean + per-test-point covariance KLLA.
//
// Pipeline (all f32 inputs; big contractions in f16 WMMA w/ f32 accumulate):
//   mean_kernel : mean = X@W + b                       (256x4)
//   jz_kernel   : JzH[(n,a), s]  = f16(prior * X@Jproj)     (1024x4096 f16)
//   pack_jx     : JxH[(b,m), s]  = f16(Jx[m,b,s])           (4096x4096 f16)
//   pack_inv    : InvH[(b,m),(c,l)] = f16(inv[b,c,m,l])     (4096x4096 f16, symmetric)
//   gemm_nt     : KzxH = JzH  x JxH^T   (WMMA f16->f32, store f16)
//   gemm_nt     : Tmp  = KzxH x InvH^T  (InvH symmetric => == KzxH x InvM)
//   combine     : per n: Kzz = Jz Jz^T, K2 = Tmp Kzx^T, out = Kzz - K2
// ---------------------------------------------------------------------------

typedef __attribute__((ext_vector_type(16))) _Float16 v16h;
typedef __attribute__((ext_vector_type(8)))  _Float16 v8h;
typedef __attribute__((ext_vector_type(8)))  float    v8f;

#define N_TEST   256
#define N_TRAIN  1024
#define OUT_DIM  4
#define D_IN     256
#define S_DIM    4096
#define P_DIM    (OUT_DIM * N_TRAIN)   // 4096 = (b,m) and (c,l) flat index
#define R_DIM    (N_TEST * OUT_DIM)    // 1024 = (n,a) flat row index

#define WMMA_F16(af, bf, cc) \
    __builtin_amdgcn_wmma_f32_16x16x32_f16(false, (af), false, (bf), (short)0, (cc), false, false)

// ---------------------------------------------------------------- mean -----
__global__ __launch_bounds__(256)
void mean_kernel(const float* __restrict__ X, const float* __restrict__ W,
                 const float* __restrict__ b, float* __restrict__ out) {
    int i = blockIdx.x * blockDim.x + threadIdx.x;   // i = n*4 + a
    if (i >= N_TEST * OUT_DIM) return;
    int n = i >> 2, a = i & 3;
    float acc = b[a];
    for (int d = 0; d < D_IN; ++d)
        acc += X[n * D_IN + d] * W[d * OUT_DIM + a];
    out[i] = acc;
}

// ------------------------------------------------------------------ Jz -----
// Jz[n, a*4096+s] = prior * sum_d X[n,d] * Jproj[d, a*4096+s]  -> f16
// grid (64, 16): x = 256-col chunk of the 16384 (a,s) columns, y = 16 n's
__global__ __launch_bounds__(256)
void jz_kernel(const float* __restrict__ X, const float* __restrict__ Jproj,
               const float* __restrict__ prior, _Float16* __restrict__ JzH) {
    __shared__ float xs[16 * D_IN];
    const int tid = threadIdx.x;
    const int colBase = blockIdx.x * 256;
    const int nBase   = blockIdx.y * 16;
    for (int idx = tid; idx < 16 * D_IN; idx += 256)
        xs[idx] = X[(nBase + (idx >> 8)) * D_IN + (idx & 255)];
    __syncthreads();

    const int col = colBase + tid;          // col in [0, 16384) = (a,s)
    float acc[16];
#pragma unroll
    for (int t = 0; t < 16; ++t) acc[t] = 0.0f;
    for (int d = 0; d < D_IN; ++d) {
        float jp = Jproj[(size_t)d * (OUT_DIM * S_DIM) + col];
#pragma unroll
        for (int t = 0; t < 16; ++t) acc[t] += xs[t * D_IN + d] * jp;
    }
    const float pr = prior[0];
#pragma unroll
    for (int t = 0; t < 16; ++t)
        JzH[(size_t)(nBase + t) * (OUT_DIM * S_DIM) + col] = (_Float16)(acc[t] * pr);
}

// ---------------------------------------------------------------- packs ----
// JxH[p= b*1024+m][s] = f16( Jx[m][b][s] ), coalesced on src and dst
__global__ __launch_bounds__(256)
void pack_jx(const float* __restrict__ Jx, _Float16* __restrict__ JxH) {
    size_t dst = (size_t)blockIdx.x * blockDim.x + threadIdx.x;
    int p = (int)(dst >> 12), s = (int)(dst & 4095);
    int b = p >> 10, m = p & 1023;
    JxH[dst] = (_Float16)Jx[((size_t)m * OUT_DIM + b) * S_DIM + s];
}

// InvH[p=(b,m)][q=(c,l)] = f16( inv[b][c][m][l] ); symmetric matrix
__global__ __launch_bounds__(256)
void pack_inv(const float* __restrict__ inv, _Float16* __restrict__ InvH) {
    size_t dst = (size_t)blockIdx.x * blockDim.x + threadIdx.x;
    int p = (int)(dst >> 12), q = (int)(dst & 4095);
    int b = p >> 10, m = p & 1023;
    int c = q >> 10, l = q & 1023;
    InvH[dst] = (_Float16)inv[(((size_t)(b * OUT_DIM + c)) << 20) + ((size_t)m << 10) + l];
}

// ----------------------------------------------------------- WMMA GEMM ----
// C[M x N] (f32 and/or f16) = A[M x K] * Bt[N x K]^T, both f16 row-major
// along K ("NT"). Block = 128 threads = 4 waves; each wave computes a
// 32x64 output tile (2 M-tiles x 4 N-tiles -> 8 v8f accumulators), with
// explicit double-buffering of the f16 fragments so next-step loads
// overlap current-step v_wmma issue (waits become partial, not 0x0).
// Fragment addressing: A and B both use contiguous k = 16*hi + s per lane half.
__global__ __launch_bounds__(128)
void gemm_nt_wmma(const _Float16* __restrict__ A, const _Float16* __restrict__ Bt,
                  float* __restrict__ Cf, _Float16* __restrict__ Ch,
                  int N, int K) {
    const int lane = threadIdx.x & 31;
    const int wave = threadIdx.x >> 5;
    const int hi   = lane >> 4;      // 0: lanes 0-15, 1: lanes 16-31
    const int lo16 = lane & 15;
    const int mBase = blockIdx.y * 128 + wave * 16;   // 2nd M-tile at +64
    const int nBase = blockIdx.x * 64;

    v8f acc[8];
#pragma unroll
    for (int i = 0; i < 8; ++i) acc[i] = (v8f){};

    const _Float16* a0p = A + (size_t)(mBase + lo16) * K;
    const _Float16* a1p = a0p + (size_t)64 * K;
    const _Float16* b0p = Bt + (size_t)(nBase +  0 + lo16) * K + hi * 16;
    const _Float16* b1p = Bt + (size_t)(nBase + 16 + lo16) * K + hi * 16;
    const _Float16* b2p = Bt + (size_t)(nBase + 32 + lo16) * K + hi * 16;
    const _Float16* b3p = Bt + (size_t)(nBase + 48 + lo16) * K + hi * 16;

    // A fragment (16x32 f16): lanes 0-15 hold K {k..k+15},
    // lanes 16-31 hold K {k+16..k+31}
    auto ldA = [&](const _Float16* p, int k) -> v16h {
        v8h lo = *(const v8h*)(p + k + hi * 16);
        v8h hh = *(const v8h*)(p + k + 8 + hi * 16);
        return __builtin_shufflevector(lo, hh,
               0, 1, 2, 3, 4, 5, 6, 7, 8, 9, 10, 11, 12, 13, 14, 15);
    };
    // B fragment (32x16): lane holds column lo16, 16 contiguous K's
    auto ldB = [&](const _Float16* p, int k) -> v16h {
        return *(const v16h*)(p + k);
    };

    v16h a0c = ldA(a0p, 0), a1c = ldA(a1p, 0);
    v16h b0c = ldB(b0p, 0), b1c = ldB(b1p, 0);
    v16h b2c = ldB(b2p, 0), b3c = ldB(b3p, 0);

    int k0 = 0;
    for (; k0 < K - 32; k0 += 32) {
        const int kn = k0 + 32;
        // issue next-buffer loads before consuming current buffer
        v16h a0n = ldA(a0p, kn), a1n = ldA(a1p, kn);
        v16h b0n = ldB(b0p, kn), b1n = ldB(b1p, kn);
        v16h b2n = ldB(b2p, kn), b3n = ldB(b3p, kn);
        __builtin_prefetch(a0p + k0 + 1024, 0, 3);
        __builtin_prefetch(b0p + k0 + 1024, 0, 3);
        __builtin_prefetch(b2p + k0 + 1024, 0, 3);

        acc[0] = WMMA_F16(a0c, b0c, acc[0]);
        acc[1] = WMMA_F16(a0c, b1c, acc[1]);
        acc[2] = WMMA_F16(a0c, b2c, acc[2]);
        acc[3] = WMMA_F16(a0c, b3c, acc[3]);
        acc[4] = WMMA_F16(a1c, b0c, acc[4]);
        acc[5] = WMMA_F16(a1c, b1c, acc[5]);
        acc[6] = WMMA_F16(a1c, b2c, acc[6]);
        acc[7] = WMMA_F16(a1c, b3c, acc[7]);

        a0c = a0n; a1c = a1n;
        b0c = b0n; b1c = b1n; b2c = b2n; b3c = b3n;
    }
    // tail (k0 == K-32): consume the last buffer
    acc[0] = WMMA_F16(a0c, b0c, acc[0]);
    acc[1] = WMMA_F16(a0c, b1c, acc[1]);
    acc[2] = WMMA_F16(a0c, b2c, acc[2]);
    acc[3] = WMMA_F16(a0c, b3c, acc[3]);
    acc[4] = WMMA_F16(a1c, b0c, acc[4]);
    acc[5] = WMMA_F16(a1c, b1c, acc[5]);
    acc[6] = WMMA_F16(a1c, b2c, acc[6]);
    acc[7] = WMMA_F16(a1c, b3c, acc[7]);

    // D layout: VGPR v -> row (tileBase + v + hi*8), col (nBase + lo16)
    const int ccol = nBase + lo16;
#pragma unroll
    for (int mt = 0; mt < 2; ++mt) {
#pragma unroll
        for (int v = 0; v < 8; ++v) {
            const size_t base = (size_t)(mBase + mt * 64 + v + hi * 8) * N + ccol;
            if (Cf) {
                Cf[base +  0] = acc[mt * 4 + 0][v];
                Cf[base + 16] = acc[mt * 4 + 1][v];
                Cf[base + 32] = acc[mt * 4 + 2][v];
                Cf[base + 48] = acc[mt * 4 + 3][v];
            }
            if (Ch) {
                Ch[base +  0] = (_Float16)acc[mt * 4 + 0][v];
                Ch[base + 16] = (_Float16)acc[mt * 4 + 1][v];
                Ch[base + 32] = (_Float16)acc[mt * 4 + 2][v];
                Ch[base + 48] = (_Float16)acc[mt * 4 + 3][v];
            }
        }
    }
}

// -------------------------------------------------------------- combine ---
// per n: Kzz[a,d] = sum_s Jz[n,a,s]Jz[n,d,s]
//        K2 [a,d] = sum_q Tmp[(n,a),q] * Kzx[(n,d),q]
//        out[1024 + n*16 + a*4 + d] = Kzz - K2
__global__ __launch_bounds__(256)
void combine_kernel(const _Float16* __restrict__ JzH,
                    const _Float16* __restrict__ KzxH,
                    const float* __restrict__ Tmp,
                    float* __restrict__ out) {
    const int n = blockIdx.x, tid = threadIdx.x;
    const _Float16* jz = JzH  + (size_t)n * OUT_DIM * S_DIM;
    const _Float16* kz = KzxH + (size_t)n * OUT_DIM * P_DIM;
    const float*    tp = Tmp  + (size_t)n * OUT_DIM * P_DIM;

    float kzz[16], k2[16];
#pragma unroll
    for (int i = 0; i < 16; ++i) { kzz[i] = 0.0f; k2[i] = 0.0f; }

    for (int q = tid; q < S_DIM; q += 256) {
        float jzv[4], kzv[4], tpv[4];
#pragma unroll
        for (int a = 0; a < 4; ++a) {
            jzv[a] = (float)jz[a * S_DIM + q];
            kzv[a] = (float)kz[a * P_DIM + q];
            tpv[a] = tp[a * P_DIM + q];
        }
#pragma unroll
        for (int a = 0; a < 4; ++a)
#pragma unroll
            for (int d = 0; d < 4; ++d) {
                kzz[a * 4 + d] += jzv[a] * jzv[d];
                k2 [a * 4 + d] += tpv[a] * kzv[d];
            }
    }
    // wave32 reduce each of the 32 sums
#pragma unroll
    for (int i = 0; i < 16; ++i)
        for (int off = 16; off > 0; off >>= 1) {
            kzz[i] += __shfl_down(kzz[i], off, 32);
            k2 [i] += __shfl_down(k2 [i], off, 32);
        }
    __shared__ float part[8][32];
    const int wid = tid >> 5, lane = tid & 31;
    if (lane == 0) {
#pragma unroll
        for (int i = 0; i < 16; ++i) {
            part[wid][i]      = kzz[i];
            part[wid][16 + i] = k2[i];
        }
    }
    __syncthreads();
    if (tid < 16) {
        float s1 = 0.0f, s2 = 0.0f;
#pragma unroll
        for (int w = 0; w < 8; ++w) { s1 += part[w][tid]; s2 += part[w][16 + tid]; }
        out[N_TEST * OUT_DIM + n * 16 + tid] = s1 - s2;
    }
}

// ---------------------------------------------------------------------------
extern "C" void kernel_launch(void* const* d_in, const int* in_sizes, int n_in,
                              void* d_out, int out_size, void* d_ws, size_t ws_size,
                              hipStream_t stream) {
    const float* X     = (const float*)d_in[0];   // (256,256)
    const float* W_net = (const float*)d_in[1];   // (256,4)
    const float* b_net = (const float*)d_in[2];   // (4,)
    const float* Jproj = (const float*)d_in[3];   // (256,4,4096)
    const float* Jx    = (const float*)d_in[4];   // (1024,4,4096)
    const float* inv   = (const float*)d_in[5];   // (4,4,1024,1024)
    const float* prior = (const float*)d_in[6];   // (1,)
    float* out = (float*)d_out;                   // 1024 mean + 4096 KLLA

    char* ws = (char*)d_ws;
    _Float16* JzH  = (_Float16*)(ws);                          //  8 MB (1024x4096)
    _Float16* JxH  = (_Float16*)(ws + ( 8ull << 20));          // 32 MB (4096x4096)
    _Float16* InvH = (_Float16*)(ws + (40ull << 20));          // 32 MB (4096x4096)
    _Float16* KzxH = (_Float16*)(ws + (72ull << 20));          //  8 MB (1024x4096)
    float*    Tmp  = (float*)   (ws + (80ull << 20));          // 16 MB (1024x4096)

    mean_kernel<<<4, 256, 0, stream>>>(X, W_net, b_net, out);
    jz_kernel<<<dim3(64, 16), 256, 0, stream>>>(X, Jproj, prior, JzH);
    pack_jx <<<(P_DIM * S_DIM) / 256, 256, 0, stream>>>(Jx, JxH);
    pack_inv<<<(P_DIM * P_DIM) / 256, 256, 0, stream>>>(inv, InvH);

    // Kzx[(n,a),(b,m)] = Jz . Jx^T   (f16 out feeds next GEMM + combine)
    gemm_nt_wmma<<<dim3(P_DIM / 64, R_DIM / 128), 128, 0, stream>>>(
        JzH, JxH, nullptr, KzxH, P_DIM, S_DIM);
    // Tmp[(n,a),(c,l)] = Kzx . InvM  (InvH symmetric -> NT form is exact)
    gemm_nt_wmma<<<dim3(P_DIM / 64, R_DIM / 128), 128, 0, stream>>>(
        KzxH, InvH, Tmp, nullptr, P_DIM, P_DIM);

    combine_kernel<<<N_TEST, 256, 0, stream>>>(JzH, KzxH, Tmp, out);
}
